// LowFreqPart_27754078667540
// MI455X (gfx1250) — matrix operand order repaired and checked by reading, not compiled, not run on hardware
//
#include <hip/hip_runtime.h>
#include <hip/hip_bf16.h>

typedef __attribute__((ext_vector_type(16))) _Float16 v16h;
typedef __attribute__((ext_vector_type(8)))  _Float16 v8h;
typedef __attribute__((ext_vector_type(8)))  float    v8f;

// ---------------------------------------------------------------------------
// small utility kernels
// ---------------------------------------------------------------------------
__global__ void init_kernel(unsigned* maxbits) {
    if (threadIdx.x == 0 && blockIdx.x == 0) maxbits[0] = 0u;
}

__global__ void max_reduce_kernel(const float* __restrict__ x, int n, unsigned* maxbits) {
    __shared__ float sm[256];
    float v = 0.0f;
    for (int j = blockIdx.x * 256 + threadIdx.x; j < n; j += gridDim.x * 256)
        v = fmaxf(v, x[j]);
    sm[threadIdx.x] = v;
    __syncthreads();
    for (int st = 128; st > 0; st >>= 1) {
        if (threadIdx.x < st) sm[threadIdx.x] = fmaxf(sm[threadIdx.x], sm[threadIdx.x + st]);
        __syncthreads();
    }
    if (threadIdx.x == 0) atomicMax(maxbits, __float_as_uint(sm[0]));
}

__global__ void finalize_scale_kernel(const unsigned* maxbits, float* inv) {
    if (threadIdx.x == 0 && blockIdx.x == 0)
        inv[0] = (__uint_as_float(maxbits[0]) > 1.0f) ? (1.0f / 255.0f) : 1.0f;
}

__global__ void cvt_f16_kernel(const float* __restrict__ in, _Float16* __restrict__ out, int n) {
    int i = blockIdx.x * blockDim.x + threadIdx.x;
    if (i < n) out[i] = (_Float16)in[i];
}

// ---------------------------------------------------------------------------
// conv1: 3->32, 3x3, stride 2, pad 1, fused BN. Direct (Cin=3 too small for WMMA)
// in: x (f32 NCHW, scaled by inv[0]);  out: f16 NHWC (16,128,96,32)
// ---------------------------------------------------------------------------
__global__ void conv1_kernel(const float* __restrict__ x, const float* __restrict__ inv,
                             const float* __restrict__ w, const float* __restrict__ b,
                             const float* __restrict__ g, const float* __restrict__ be,
                             _Float16* __restrict__ out) {
    const int Hout = 128, Wout = 96, Hin = 256, Win = 192;
    int idx = blockIdx.x * blockDim.x + threadIdx.x;   // NHWC flat index, co fastest
    const int total = 16 * Hout * Wout * 32;
    if (idx >= total) return;
    int co = idx & 31;
    int r  = idx >> 5;            // n*Hout*Wout + ho*Wout + wo
    int n  = r / (Hout * Wout);
    int r2 = r - n * (Hout * Wout);
    int ho = r2 / Wout;
    int wo = r2 - ho * Wout;
    float s = inv[0];
    float acc = b[co];
    for (int ci = 0; ci < 3; ++ci)
        for (int kh = 0; kh < 3; ++kh) {
            int hi = 2 * ho - 1 + kh;
            if (hi < 0 || hi >= Hin) continue;
            for (int kw = 0; kw < 3; ++kw) {
                int wi = 2 * wo - 1 + kw;
                if (wi < 0 || wi >= Win) continue;
                acc += x[((n * 3 + ci) * Hin + hi) * Win + wi] * s * w[((co * 3 + ci) * 3 + kh) * 3 + kw];
            }
        }
    float sc = g[co] * rsqrtf(1.0f + 1e-5f);
    out[idx] = (_Float16)(acc * sc + be[co]);
}

// ---------------------------------------------------------------------------
// WMMA implicit-GEMM conv: 3x3 stride-2 pad-1, NHWC f16 activations.
// One wave per 16-pixel x 16-Cout tile; 9 taps x (Cin/32) chunks of
// v_wmma_f32_16x16x32_f16. A loaded as two aligned b128 spans per lane
// (NHWC channels contiguous == ISA A-fragment runs); B pre-swizzled into
// per-lane fragment order in LDS so each lane reads 32 contiguous bytes.
// ---------------------------------------------------------------------------
__global__ __launch_bounds__(128)
void conv3x3_s2_wmma(const _Float16* __restrict__ in,   // [N,Hin,Win,Cin]
                     const _Float16* __restrict__ wt,   // [Cout,Cin,3,3] f16 (OIHW)
                     const float* __restrict__ bias,
                     const float* __restrict__ gamma,
                     const float* __restrict__ beta,
                     _Float16* __restrict__ out,        // [N,Hout,Wout,Cout]
                     int N, int Cin, int Hin, int Win,
                     int Cout, int Hout, int Wout) {
    extern __shared__ _Float16 lds_w[];   // [(tap*nChunks+ch)*32 + lane][16]
    const int lane = threadIdx.x;         // 0..31
    const int wave = threadIdx.y;         // 0..3
    const int nChunks = Cin >> 5;
    const int totalPix = N * Hout * Wout;
    const int coBase = blockIdx.y * 16;

    // stage weights in B-fragment layout: element e of lane L in block (t,chunk)
    // holds wt[co = coBase + (L&15), c = chunk*32 + (L>=16?8:0) + (e<8?e:e+8), tap t]
    const int tid = wave * 32 + lane;
    const int wElems = 9 * Cin * 16;              // == 9*nChunks*32*16
    for (int idx = tid; idx < wElems; idx += 128) {
        int t    = idx / (Cin * 16);
        int rem  = idx - t * (Cin * 16);          // chunk*512 + l*16 + e
        int chnk = rem >> 9;
        int l    = (rem >> 4) & 31;
        int e    = rem & 15;
        int c    = (chnk << 5) + ((l >= 16) ? 8 : 0) + ((e < 8) ? e : e + 8);
        lds_w[((t * nChunks + chnk) * 32 + l) * 16 + e] =
            wt[((coBase + (l & 15)) * Cin + c) * 9 + t];
    }
    __syncthreads();

    const int tile = blockIdx.x * 4 + wave;
    const int m    = lane & 15;
    const int hiK  = (lane >> 4) ? 8 : 0;   // A-fragment K offset per lane half

    int p = tile * 16 + m;
    bool pvalid = p < totalPix;
    int pc = pvalid ? p : 0;
    int n_img = pc / (Hout * Wout);
    int r2 = pc - n_img * (Hout * Wout);
    int ho = r2 / Wout;
    int wo = r2 - ho * Wout;

    v8f acc = {};
    const v8h zero8 = {};
    for (int t = 0; t < 9; ++t) {
        int kh = t / 3, kw = t - kh * 3;
        int hi = 2 * ho - 1 + kh;
        int wi = 2 * wo - 1 + kw;
        bool svalid = pvalid && hi >= 0 && hi < Hin && wi >= 0 && wi < Win;
        int hic = min(max(hi, 0), Hin - 1);
        int wic = min(max(wi, 0), Win - 1);
        const _Float16* pix = in + (((long)n_img * Hin + hic) * Win + wic) * Cin;
        for (int ch = 0; ch < nChunks; ++ch) {
            const _Float16* p0 = pix + (ch << 5) + hiK;
            v8h lo = *(const v8h*)p0;            // k = hiK + 0..7
            v8h hi8 = *(const v8h*)(p0 + 16);    // k = 16 + hiK + 0..7
            union { v16h v; v8h h[2]; } au;
            au.h[0] = svalid ? lo : zero8;
            au.h[1] = svalid ? hi8 : zero8;
            v16h bfrag = *(const v16h*)(lds_w + (((t * nChunks + ch) * 32 + lane) << 4));
            acc = __builtin_amdgcn_wmma_f32_16x16x32_f16(
                false, au.v, false, bfrag, (short)0, acc, false, false);
        }
    }

    // epilogue: bias + optional BN, per ISA f32 C/D layout; NHWC store
    int co = coBase + (lane & 15);
    float s = 1.0f, sh = 0.0f;
    if (gamma) { s = gamma[co] * rsqrtf(1.0f + 1e-5f); sh = beta[co]; }
    float bv = bias[co];
#pragma unroll
    for (int r = 0; r < 8; ++r) {
        int mm = r + ((lane >> 4) ? 8 : 0);
        int pp = tile * 16 + mm;
        if (pp < totalPix) {
            float v = (acc[r] + bv) * s + sh;
            out[(long)pp * Cout + co] = (_Float16)v;
        }
    }
}

// ---------------------------------------------------------------------------
// conv6: 128ch 8x6 VALID -> params[n, co] (co<8). One block per (n, co).
// a5 is NHWC: [16, 8, 6, 128]
// ---------------------------------------------------------------------------
__global__ void conv6_kernel(const _Float16* __restrict__ a5, const float* __restrict__ w6,
                             const float* __restrict__ b6, float* __restrict__ params) {
    __shared__ float red[256];
    int n = blockIdx.x, co = blockIdx.y;
    float s = 0.0f;
    for (int i = threadIdx.x; i < 128 * 48; i += 256) {
        int c  = i & 127;
        int sp = i >> 7;                 // kh*6+kw
        s += (float)a5[(n * 48 + sp) * 128 + c] * w6[(co * 128 + c) * 48 + sp];
    }
    red[threadIdx.x] = s;
    __syncthreads();
    for (int st = 128; st > 0; st >>= 1) {
        if (threadIdx.x < st) red[threadIdx.x] += red[threadIdx.x + st];
        __syncthreads();
    }
    if (threadIdx.x == 0) params[n * 8 + co] = red[0] + b6[co];
}

// ---------------------------------------------------------------------------
// exposed = x * inv_scale * params[n,0]   (f32 NCHW)
// ---------------------------------------------------------------------------
__global__ void exposure_kernel(const float* __restrict__ x, const float* __restrict__ inv,
                                const float* __restrict__ params, float* __restrict__ exposed) {
    const int per = 3 * 256 * 192;
    int idx = blockIdx.x * blockDim.x + threadIdx.x;
    if (idx >= 16 * per) return;
    int n = idx / per;
    exposed[idx] = x[idx] * inv[0] * params[n * 8 + 0];
}

// ---------------------------------------------------------------------------
// local-gamma 3->3 3x3 pad1 conv; optional skip add; optional final
// gamma = expmul * sigmoid(.) fusion.
// ---------------------------------------------------------------------------
__global__ void lg_conv_kernel(const float* __restrict__ in, const float* __restrict__ w,
                               const float* __restrict__ bias, const float* __restrict__ skip,
                               const float* __restrict__ expmul, float* __restrict__ out,
                               int N, int H, int W) {
    int idx = blockIdx.x * blockDim.x + threadIdx.x;
    int total = N * H * W;
    if (idx >= total) return;
    int n = idx / (H * W);
    int r = idx - n * (H * W);
    int h = r / W;
    int wq = r - h * W;
    const int HW = H * W;
    const float* base = in + n * 3 * HW;
    float acc[3];
    for (int co = 0; co < 3; ++co) acc[co] = bias[co];
    for (int ci = 0; ci < 3; ++ci)
        for (int kh = 0; kh < 3; ++kh) {
            int hh = h - 1 + kh;
            if (hh < 0 || hh >= H) continue;
            for (int kw = 0; kw < 3; ++kw) {
                int ww = wq - 1 + kw;
                if (ww < 0 || ww >= W) continue;
                float v = base[ci * HW + hh * W + ww];
                for (int co = 0; co < 3; ++co)
                    acc[co] += v * w[((co * 3 + ci) * 3 + kh) * 3 + kw];
            }
        }
    for (int co = 0; co < 3; ++co) {
        int oidx = (n * 3 + co) * HW + h * W + wq;
        float v = acc[co];
        if (skip)   v += skip[oidx];
        if (expmul) v = expmul[oidx] * (1.0f / (1.0f + __expf(-v)));
        out[oidx] = v;
    }
}

// ---------------------------------------------------------------------------
// 5x5 bilateral, edge padding, per-(n,c) sigma_space/sigma_color from params
// ---------------------------------------------------------------------------
__global__ void bilateral_kernel(const float* __restrict__ img, const float* __restrict__ params,
                                 float* __restrict__ out, int N, int H, int W) {
    int idx = blockIdx.x * blockDim.x + threadIdx.x;
    int total = N * 3 * H * W;
    if (idx >= total) return;
    int n  = idx / (3 * H * W);
    int r  = idx - n * (3 * H * W);
    int c  = r / (H * W);
    int r2 = r - c * (H * W);
    int h  = r2 / W;
    int w  = r2 - h * W;
    const float* ch = img + (n * 3 + c) * H * W;
    float center = ch[h * W + w];
    float ss = params[n * 8 + 2 + c];
    float sc = params[n * 8 + 5 + c];
    float inv2ss = -1.0f / (2.0f * ss * ss);
    float inv2sc = -1.0f / (2.0f * sc * sc + 1e-6f);
    float num = 0.0f, den = 0.0f;
    for (int dy = -2; dy <= 2; ++dy) {
        int hh = min(max(h + dy, 0), H - 1);
        for (int dx = -2; dx <= 2; ++dx) {
            int ww = min(max(w + dx, 0), W - 1);
            float v = ch[hh * W + ww];
            float d = v - center;
            float wt = __expf((float)(dy * dy + dx * dx) * inv2ss) * __expf(d * d * inv2sc);
            num += v * wt;
            den += wt;
        }
    }
    out[idx] = num / fmaxf(den, 1e-6f);
}

// ---------------------------------------------------------------------------
// launcher
// ---------------------------------------------------------------------------
extern "C" void kernel_launch(void* const* d_in, const int* in_sizes, int n_in,
                              void* d_out, int out_size, void* d_ws, size_t ws_size,
                              hipStream_t stream) {
    (void)in_sizes; (void)n_in; (void)out_size; (void)ws_size;
    // input order: x, opp{w1,b1,g1,be1,w2,b2,g2,be2,w3,b3,g3,be3,w4,b4,w5,b5,w6,b6}, lg{w1,b1..w6,b6}
    const float* x   = (const float*)d_in[0];
    const float* w1  = (const float*)d_in[1];   const float* b1 = (const float*)d_in[2];
    const float* g1  = (const float*)d_in[3];   const float* be1 = (const float*)d_in[4];
    const float* w2  = (const float*)d_in[5];   const float* b2 = (const float*)d_in[6];
    const float* g2  = (const float*)d_in[7];   const float* be2 = (const float*)d_in[8];
    const float* w3  = (const float*)d_in[9];   const float* b3 = (const float*)d_in[10];
    const float* g3  = (const float*)d_in[11];  const float* be3 = (const float*)d_in[12];
    const float* w4  = (const float*)d_in[13];  const float* b4 = (const float*)d_in[14];
    const float* w5  = (const float*)d_in[15];  const float* b5 = (const float*)d_in[16];
    const float* w6  = (const float*)d_in[17];  const float* b6 = (const float*)d_in[18];
    const float* lgw[6]; const float* lgb[6];
    for (int i = 0; i < 6; ++i) { lgw[i] = (const float*)d_in[19 + 2 * i]; lgb[i] = (const float*)d_in[20 + 2 * i]; }

    char* ws = (char*)d_ws;
    size_t off = 0;
    auto alloc = [&](size_t bytes) -> void* {
        off = (off + 255) & ~(size_t)255;
        void* p = ws + off;
        off += bytes;
        return p;
    };

    unsigned* maxbits = (unsigned*)alloc(4);
    float*    invs    = (float*)alloc(4);
    _Float16* w2h = (_Float16*)alloc((size_t)64 * 32 * 9 * 2);
    _Float16* w3h = (_Float16*)alloc((size_t)128 * 64 * 9 * 2);
    _Float16* w4h = (_Float16*)alloc((size_t)128 * 128 * 9 * 2);
    _Float16* w5h = (_Float16*)alloc((size_t)128 * 128 * 9 * 2);
    _Float16* a1  = (_Float16*)alloc((size_t)16 * 32 * 128 * 96 * 2);   // NHWC
    _Float16* a2  = (_Float16*)alloc((size_t)16 * 64 * 64 * 48 * 2);    // NHWC
    _Float16* a3  = (_Float16*)alloc((size_t)16 * 128 * 32 * 24 * 2);   // NHWC
    _Float16* a4  = (_Float16*)alloc((size_t)16 * 128 * 16 * 12 * 2);   // NHWC
    _Float16* a5  = (_Float16*)alloc((size_t)16 * 128 * 8 * 6 * 2);     // NHWC
    float*    prm = (float*)alloc((size_t)16 * 8 * 4);
    const size_t imgElems = (size_t)16 * 3 * 256 * 192;
    float* exposed = (float*)alloc(imgElems * 4);
    float* ta = (float*)alloc(imgElems * 4);
    float* tb = (float*)alloc(imgElems * 4);
    float* tc = (float*)alloc(imgElems * 4);
    float* td = (float*)alloc(imgElems * 4);
    float* te = (float*)alloc(imgElems * 4);
    float* tg = (float*)alloc(imgElems * 4);

    const int nImg = (int)imgElems;

    // 1) input-range scale  where(max(x)>1, x/255, x)
    init_kernel<<<1, 32, 0, stream>>>(maxbits);
    max_reduce_kernel<<<256, 256, 0, stream>>>(x, nImg, maxbits);
    finalize_scale_kernel<<<1, 32, 0, stream>>>(maxbits, invs);

    // 2) weights -> f16 for WMMA convs
    cvt_f16_kernel<<<(64 * 32 * 9 + 255) / 256, 256, 0, stream>>>(w2, w2h, 64 * 32 * 9);
    cvt_f16_kernel<<<(128 * 64 * 9 + 255) / 256, 256, 0, stream>>>(w3, w3h, 128 * 64 * 9);
    cvt_f16_kernel<<<(128 * 128 * 9 + 255) / 256, 256, 0, stream>>>(w4, w4h, 128 * 128 * 9);
    cvt_f16_kernel<<<(128 * 128 * 9 + 255) / 256, 256, 0, stream>>>(w5, w5h, 128 * 128 * 9);

    // 3) opp trunk (activations NHWC f16)
    {
        int total = 16 * 32 * 128 * 96;
        conv1_kernel<<<(total + 255) / 256, 256, 0, stream>>>(x, invs, w1, b1, g1, be1, a1);
    }
    auto launch_wmma = [&](const _Float16* in, const _Float16* wt, const float* bias,
                           const float* gamma, const float* beta, _Float16* out,
                           int N, int Cin, int Hin, int Win, int Cout, int Hout, int Wout) {
        int totalPix = N * Hout * Wout;
        int tiles = (totalPix + 15) / 16;
        dim3 grid((tiles + 3) / 4, Cout / 16);
        size_t smem = (size_t)9 * Cin * 16 * sizeof(_Float16);
        conv3x3_s2_wmma<<<grid, dim3(32, 4), smem, stream>>>(
            in, wt, bias, gamma, beta, out, N, Cin, Hin, Win, Cout, Hout, Wout);
    };
    launch_wmma(a1, w2h, b2, g2, be2, a2, 16, 32, 128, 96, 64, 64, 48);
    launch_wmma(a2, w3h, b3, g3, be3, a3, 16, 64, 64, 48, 128, 32, 24);
    launch_wmma(a3, w4h, b4, nullptr, nullptr, a4, 16, 128, 32, 24, 128, 16, 12);
    launch_wmma(a4, w5h, b5, nullptr, nullptr, a5, 16, 128, 16, 12, 128, 8, 6);
    conv6_kernel<<<dim3(16, 8), 256, 0, stream>>>(a5, w6, b6, prm);

    // 4) exposure + local gamma chain (f32 NCHW)
    exposure_kernel<<<(nImg + 255) / 256, 256, 0, stream>>>(x, invs, prm, exposed);
    int pixTotal = 16 * 256 * 192;
    int pgrid = (pixTotal + 255) / 256;
    lg_conv_kernel<<<pgrid, 256, 0, stream>>>(exposed, lgw[0], lgb[0], nullptr, nullptr, ta, 16, 256, 192);
    lg_conv_kernel<<<pgrid, 256, 0, stream>>>(ta, lgw[1], lgb[1], nullptr, nullptr, tb, 16, 256, 192);
    lg_conv_kernel<<<pgrid, 256, 0, stream>>>(tb, lgw[2], lgb[2], nullptr, nullptr, tc, 16, 256, 192);
    lg_conv_kernel<<<pgrid, 256, 0, stream>>>(tc, lgw[3], lgb[3], tc, nullptr, td, 16, 256, 192);
    lg_conv_kernel<<<pgrid, 256, 0, stream>>>(td, lgw[4], lgb[4], tb, nullptr, te, 16, 256, 192);
    // final: gamma_img = exposed * sigmoid(conv(te)+ta)
    lg_conv_kernel<<<pgrid, 256, 0, stream>>>(te, lgw[5], lgb[5], ta, exposed, tg, 16, 256, 192);

    // 5) bilateral -> output
    bilateral_kernel<<<(nImg + 255) / 256, 256, 0, stream>>>(tg, prm, (float*)d_out, 16, 256, 192);
}
